// DetNet_51342039056709
// MI455X (gfx1250) — matrix operand order repaired and checked by reading, not compiled, not run on hardware
//
#include <hip/hip_runtime.h>
#include <hip/hip_bf16.h>

// ---------------------------------------------------------------------------
// DetNet on MI455X (gfx1250): bf16 WMMA GEMMs + LDS-staged per-sample matvecs.
//
// Sizes: L=30, B=64, S=72 (2S=144), OH=4, VL=128 (2VL=256), ZL=512 (2ZL=1024),
// BATCH=2048.  K of W1 padded 400 -> 416 (13 * 32).
//
// Workspace layout (assumed ws_size >= ~220 MB):
//   psi_e bf16, psi_o bf16, W1/W2/W3 bf16 (W1 K-padded), x/v/xoh f32 carries,
//   qv bf16 (2048 x 416).
// ---------------------------------------------------------------------------

#define LAYERS 30
#define NBATCH 2048
#define TWO_S 144
#define TWO_B 128
#define TWO_V 256
#define TWO_Z 1024
#define SOH   288
#define K1PAD 416   // 400 padded to multiple of 32

// Async global->LDS copy path (gfx1250): guarded so absence of the builtins
// falls back to the plain VGPR round-trip copy.
#if defined(__gfx1250__) && \
    __has_builtin(__builtin_amdgcn_global_load_async_to_lds_b128) && \
    __has_builtin(__builtin_amdgcn_s_wait_asynccnt)
#define HAVE_ASYNC_LDS 1
#else
#define HAVE_ASYNC_LDS 0
#endif

typedef float  v8f   __attribute__((ext_vector_type(8)));
typedef __bf16 v16bf __attribute__((ext_vector_type(16)));
typedef int    v4i   __attribute__((vector_size(16)));   // matches builtin param

#if HAVE_ASYNC_LDS
typedef __attribute__((address_space(1))) v4i* g_v4i_ptr;
typedef __attribute__((address_space(3))) v4i* l_v4i_ptr;
#endif

struct bfpack { uint4 lo, hi; };

// A-operand (16x32 bf16): lane m (m=lane&15, h=lane>>4) holds row m,
// halves 0..15 = K[16h .. 16h+15] (contiguous 32 B).
// p must already point at row_base + 16*h; chunks at +0 and +8 elements.
__device__ __forceinline__ v16bf load_opA(const __bf16* p) {
    bfpack s;
    s.lo = *(const uint4*)(p);
    s.hi = *(const uint4*)(p + 8);
    return __builtin_bit_cast(v16bf, s);
}

// B-operand (32x16 bf16): lane n (n=lane&15, h=lane>>4) holds column n
// (= weight row n), halves 0..15 = K[16h .. 16h+15] (contiguous 32 B).
__device__ __forceinline__ v16bf load_opB(const __bf16* p) {
    bfpack s;
    s.lo = *(const uint4*)(p);
    s.hi = *(const uint4*)(p + 8);
    return __builtin_bit_cast(v16bf, s);
}

__device__ __forceinline__ v8f wmma_bf16(v16bf a, v16bf b, v8f c) {
    // D = A x B + C, 16x16x32 bf16 -> f32
    return __builtin_amdgcn_wmma_f32_16x16x32_bf16(false, a, false, b,
                                                   (short)0, c, false, false);
}

// ---------------------------------------------------------------------------
// Converters / init
// ---------------------------------------------------------------------------
__global__ __launch_bounds__(256) void conv_f32_bf16(const float* __restrict__ in,
                                                     __bf16* __restrict__ out, long n) {
    long i = (long)blockIdx.x * blockDim.x + threadIdx.x;
    long stride = (long)gridDim.x * blockDim.x;
    for (; i < n; i += stride) out[i] = (__bf16)in[i];
}

__global__ __launch_bounds__(256) void conv_w1_pad(const float* __restrict__ in,
                                                   __bf16* __restrict__ out, long n) {
    long i = (long)blockIdx.x * blockDim.x + threadIdx.x;
    long stride = (long)gridDim.x * blockDim.x;
    for (; i < n; i += stride) {
        long r = i / K1PAD;
        int  k = (int)(i % K1PAD);
        out[i] = (k < 400) ? (__bf16)in[r * 400 + k] : (__bf16)0.f;
    }
}

__global__ __launch_bounds__(256) void zero_f32(float* __restrict__ p, long n) {
    long i = (long)blockIdx.x * blockDim.x + threadIdx.x;
    long stride = (long)gridDim.x * blockDim.x;
    for (; i < n; i += stride) p[i] = 0.f;
}

// ---------------------------------------------------------------------------
// layerA: per-sample Psi work -> q, and pack qv = bf16([q; v; pad]) (len 416)
// grid = 2048 blocks x 128 threads; Psi tile staged in LDS (bf16, 36 KB)
// ---------------------------------------------------------------------------
__global__ __launch_bounds__(128) void detnet_layerA(
    const float* __restrict__ y_e, const float* __restrict__ y_o,
    const __bf16* __restrict__ psi_e, const __bf16* __restrict__ psi_o,
    const float* __restrict__ x_ws, const float* __restrict__ v_ws,
    __bf16* __restrict__ qv,
    const float* __restrict__ d1, const float* __restrict__ d2,
    const float* __restrict__ d4, int layer)
{
    __shared__ __bf16 psi_s[TWO_B * TWO_S];   // 128*144 bf16 = 36864 B
    __shared__ float  x_s[TWO_S];
    __shared__ float  px_s[TWO_B];
    __shared__ float  w_s[64];
    __shared__ float  q_s[TWO_S];

    const int b = blockIdx.x;
    const int t = threadIdx.x;
    const float d1l = d1[layer], d2l = d2[layer], d4l = d4[layer];

    for (int s = t; s < TWO_S; s += 128) {
        float xv = x_ws[(size_t)b * TWO_S + s];
        x_s[s] = xv;
        q_s[s] = xv;
    }

    for (int pass = 0; pass < 2; ++pass) {
        const __bf16* src = (pass == 0 ? psi_e : psi_o) + (size_t)b * (TWO_B * TWO_S);
        __syncthreads();
        // stage Psi[b] into LDS (16 B chunks; 2304 total, 18 per thread)
#if HAVE_ASYNC_LDS
        for (int i = t; i < (TWO_B * TWO_S) / 8; i += 128) {
            __builtin_amdgcn_global_load_async_to_lds_b128(
                (g_v4i_ptr)(src + (size_t)i * 8),
                (l_v4i_ptr)(psi_s + (size_t)i * 8),
                0, 0);
        }
        __builtin_amdgcn_s_wait_asynccnt(0);
#else
        for (int i = t; i < (TWO_B * TWO_S) / 8; i += 128)
            ((uint4*)psi_s)[i] = ((const uint4*)src)[i];
#endif
        __syncthreads();

        // row pass: P_x[m] = sum_s Psi[m,s] * x[s]
        float acc = 0.f;
        const __bf16* row = &psi_s[t * TWO_S];
        for (int s = 0; s < TWO_S; ++s) acc += (float)row[s] * x_s[s];
        px_s[t] = acc;
        __syncthreads();

        // combined weight vector: w_e = -d1*y_e + d2*sql_e ; w_o = -d2*y_o + d4*sql_o
        if (t < 64) {
            float p0 = px_s[t], p1 = px_s[t + 64];
            float sql = p0 * p0 + p1 * p1;
            float yv  = (pass == 0 ? y_e : y_o)[(size_t)b * 64 + t];
            w_s[t] = (pass == 0) ? (-d1l * yv + d2l * sql)
                                 : (-d2l * yv + d4l * sql);
        }
        __syncthreads();
        float u = px_s[t] * w_s[t & 63];
        px_s[t] = u;                 // reuse as u[m]
        __syncthreads();

        // column pass: q[s] += sum_m Psi[m,s] * u[m]
        for (int s = t; s < TWO_S; s += 128) {
            float a2 = 0.f;
            for (int m = 0; m < TWO_B; ++m)
                a2 += (float)psi_s[m * TWO_S + s] * px_s[m];
            q_s[s] += a2;
        }
    }
    __syncthreads();

    __bf16* qrow = qv + (size_t)b * K1PAD;
    for (int s = t; s < TWO_S; s += 128) qrow[s] = (__bf16)q_s[s];
    for (int j = t; j < TWO_V; j += 128)
        qrow[TWO_S + j] = (__bf16)v_ws[(size_t)b * TWO_V + j];
    if (t < K1PAD - 400) qrow[400 + t] = (__bf16)0.f;
}

// ---------------------------------------------------------------------------
// layerB: the three GEMMs, fused per 32-row batch tile.
//   z  = relu(QV[32x416] @ W1^T + b1)  -> LDS (bf16, 32x1024 = 64 KB)
//   xoh += z @ W2^T + b2               (stride 288)
//   v   += z @ W3^T + b3               (stride 256)
// grid = 64 blocks x 256 threads (8 waves)
// ---------------------------------------------------------------------------
__global__ __launch_bounds__(256) void detnet_layerB(
    const __bf16* __restrict__ qv,
    const __bf16* __restrict__ w1, const float* __restrict__ b1,
    const __bf16* __restrict__ w2, const float* __restrict__ b2,
    const __bf16* __restrict__ w3, const float* __restrict__ b3,
    float* __restrict__ xoh_ws, float* __restrict__ v_ws)
{
    __shared__ __bf16 zt[32 * TWO_Z];          // 65536 B

    const int row0 = blockIdx.x * 32;
    const int wave = threadIdx.x >> 5;
    const int lane = threadIdx.x & 31;
    const int l15  = lane & 15;
    const int nh   = lane >> 4;

    // ---- stage 1: z = relu(QV @ W1^T + b1), K = 416 ----
    {
        const int rb   = wave & 1;             // 2 row blocks of 16
        const int cgrp = wave >> 1;            // 4 column groups of 16 cb each
        const __bf16* aBase =
            qv + (size_t)(row0 + rb * 16 + l15) * K1PAD + 16 * nh;

        for (int h = 0; h < 2; ++h) {
            const int cb0 = cgrp * 16 + h * 8;

            // hoist the 8 weight-row base pointers out of the k loop
            const __bf16* bptr[8];
            #pragma unroll
            for (int j = 0; j < 8; ++j)
                bptr[j] = w1 + (size_t)((cb0 + j) * 16 + l15) * K1PAD + 16 * nh;

            v8f acc[8];
            #pragma unroll
            for (int j = 0; j < 8; ++j)
                acc[j] = (v8f){0.f, 0.f, 0.f, 0.f, 0.f, 0.f, 0.f, 0.f};

            for (int k = 0; k < K1PAD; k += 32) {
                // issue ALL operand loads first (distinct registers) so the
                // compiler can clause them and overlap them with the WMMAs
                v16bf a = load_opA(aBase + k);
                v16bf bm[8];
                #pragma unroll
                for (int j = 0; j < 8; ++j) bm[j] = load_opB(bptr[j] + k);

                // prefetch next k-step of the streaming operands
                __builtin_prefetch(aBase + k + 32, 0, 1);
                __builtin_prefetch(bptr[0] + k + 32, 0, 1);

                #pragma unroll
                for (int j = 0; j < 8; ++j)
                    acc[j] = wmma_bf16(a, bm[j], acc[j]);
            }

            #pragma unroll
            for (int j = 0; j < 8; ++j) {
                const int n = (cb0 + j) * 16 + l15;
                const float bias = b1[n];
                #pragma unroll
                for (int i = 0; i < 8; ++i) {
                    const int m = rb * 16 + nh * 8 + i;
                    float z = acc[j][i] + bias;
                    zt[m * TWO_Z + n] = (__bf16)fmaxf(z, 0.f);
                }
            }
        }
    }
    __syncthreads();

    // ---- stage 2: xoh += z @ W2^T + b2  (36 tiles: 2 x 18) ----
    for (int tt = wave; tt < 36; tt += 8) {
        const int rb = tt & 1, cb = tt >> 1;
        const __bf16* aBase = &zt[(rb * 16 + l15) * TWO_Z + 16 * nh];
        const __bf16* bBase = w2 + (size_t)(cb * 16 + l15) * TWO_Z + 16 * nh;
        v8f acc = (v8f){0.f, 0.f, 0.f, 0.f, 0.f, 0.f, 0.f, 0.f};
        for (int k = 0; k < TWO_Z; k += 64) {      // unroll x2, distinct regs
            v16bf a0 = load_opA(aBase + k);
            v16bf b0 = load_opB(bBase + k);
            v16bf a1 = load_opA(aBase + k + 32);
            v16bf b1v = load_opB(bBase + k + 32);
            __builtin_prefetch(bBase + k + 64, 0, 1);
            acc = wmma_bf16(a0, b0, acc);
            acc = wmma_bf16(a1, b1v, acc);
        }
        const int n = cb * 16 + l15;
        const float bias = b2[n];
        #pragma unroll
        for (int i = 0; i < 8; ++i) {
            const int row = row0 + rb * 16 + nh * 8 + i;
            const size_t idx = (size_t)row * SOH + n;
            xoh_ws[idx] = xoh_ws[idx] + acc[i] + bias;
        }
    }

    // ---- stage 3: v += z @ W3^T + b3  (32 tiles: 2 x 16) ----
    for (int tt = wave; tt < 32; tt += 8) {
        const int rb = tt & 1, cb = tt >> 1;
        const __bf16* aBase = &zt[(rb * 16 + l15) * TWO_Z + 16 * nh];
        const __bf16* bBase = w3 + (size_t)(cb * 16 + l15) * TWO_Z + 16 * nh;
        v8f acc = (v8f){0.f, 0.f, 0.f, 0.f, 0.f, 0.f, 0.f, 0.f};
        for (int k = 0; k < TWO_Z; k += 64) {      // unroll x2, distinct regs
            v16bf a0 = load_opA(aBase + k);
            v16bf b0 = load_opB(bBase + k);
            v16bf a1 = load_opA(aBase + k + 32);
            v16bf b1v = load_opB(bBase + k + 32);
            __builtin_prefetch(bBase + k + 64, 0, 1);
            acc = wmma_bf16(a0, b0, acc);
            acc = wmma_bf16(a1, b1v, acc);
        }
        const int n = cb * 16 + l15;
        const float bias = b3[n];
        #pragma unroll
        for (int i = 0; i < 8; ++i) {
            const int row = row0 + rb * 16 + nh * 8 + i;
            const size_t idx = (size_t)row * TWO_V + n;
            v_ws[idx] = v_ws[idx] + acc[i] + bias;
        }
    }
}

// ---------------------------------------------------------------------------
// layerC: softmax over OH=4 bins, map to constellation, emit x and outputs.
// grid = 2048 blocks x 128 threads
// ---------------------------------------------------------------------------
__global__ __launch_bounds__(128) void detnet_layerC(
    const float* __restrict__ xoh_ws,
    const float* __restrict__ mapp_re, const float* __restrict__ mapp_im,
    float* __restrict__ x_ws, float* __restrict__ out_xs,
    float* __restrict__ out_xoh, int layer)
{
    const int b = blockIdx.x;
    const int t = threadIdx.x;
    const float* row = xoh_ws + (size_t)b * SOH;
    float* oxoh = out_xoh + ((size_t)layer * NBATCH + b) * SOH;
    for (int i = t; i < SOH; i += 128) oxoh[i] = row[i];

    if (t < 72) {
        float a0 = row[4 * t + 0], a1 = row[4 * t + 1];
        float a2 = row[4 * t + 2], a3 = row[4 * t + 3];
        float m  = fmaxf(fmaxf(a0, a1), fmaxf(a2, a3));
        float e0 = __expf(a0 - m), e1 = __expf(a1 - m);
        float e2 = __expf(a2 - m), e3 = __expf(a3 - m);
        float inv = 1.f / (e0 + e1 + e2 + e3);
        float xr = (e0 * mapp_re[0] + e1 * mapp_re[1] +
                    e2 * mapp_re[2] + e3 * mapp_re[3]) * inv;
        float xi = (e0 * mapp_im[0] + e1 * mapp_im[1] +
                    e2 * mapp_im[2] + e3 * mapp_im[3]) * inv;
        x_ws[(size_t)b * TWO_S + t]      = xr;
        x_ws[(size_t)b * TWO_S + 72 + t] = xi;
        float* oxs = out_xs + ((size_t)layer * NBATCH + b) * TWO_S;
        oxs[t]      = xr;
        oxs[72 + t] = xi;
    }
}

// ---------------------------------------------------------------------------
extern "C" void kernel_launch(void* const* d_in, const int* in_sizes, int n_in,
                              void* d_out, int out_size, void* d_ws, size_t ws_size,
                              hipStream_t stream) {
    (void)in_sizes; (void)n_in; (void)out_size; (void)ws_size;

    const float* y_e     = (const float*)d_in[0];
    const float* y_o     = (const float*)d_in[1];
    const float* Psi_e   = (const float*)d_in[2];
    const float* Psi_o   = (const float*)d_in[3];
    const float* mapp_re = (const float*)d_in[4];
    const float* mapp_im = (const float*)d_in[5];
    const float* W1      = (const float*)d_in[6];
    const float* b1      = (const float*)d_in[7];
    const float* W2      = (const float*)d_in[8];
    const float* b2      = (const float*)d_in[9];
    const float* W3      = (const float*)d_in[10];
    const float* b3      = (const float*)d_in[11];
    const float* d1      = (const float*)d_in[12];
    const float* d2      = (const float*)d_in[13];
    const float* d4      = (const float*)d_in[14];

    float* out_xs  = (float*)d_out;
    float* out_xoh = out_xs + (size_t)LAYERS * NBATCH * TWO_S;

    // carve workspace (all region sizes are multiples of 256 bytes)
    char*  ws  = (char*)d_ws;
    size_t off = 0;
    auto carve = [&](size_t bytes) -> void* {
        void* p = ws + off;
        off += (bytes + 255) & ~(size_t)255;
        return p;
    };
    const size_t psiElems = (size_t)NBATCH * TWO_B * TWO_S;
    __bf16* psi_e_b = (__bf16*)carve(psiElems * 2);
    __bf16* psi_o_b = (__bf16*)carve(psiElems * 2);
    __bf16* w1b = (__bf16*)carve((size_t)LAYERS * TWO_Z * K1PAD * 2);
    __bf16* w2b = (__bf16*)carve((size_t)LAYERS * SOH   * TWO_Z * 2);
    __bf16* w3b = (__bf16*)carve((size_t)LAYERS * TWO_V * TWO_Z * 2);
    float*  x_ws   = (float*)carve((size_t)NBATCH * TWO_S * 4);
    float*  v_ws   = (float*)carve((size_t)NBATCH * TWO_V * 4);
    float*  xoh_ws = (float*)carve((size_t)NBATCH * SOH * 4);
    __bf16* qv_b   = (__bf16*)carve((size_t)NBATCH * K1PAD * 2);

    // one-time conversions (run every call; deterministic)
    conv_f32_bf16<<<2048, 256, 0, stream>>>(Psi_e, psi_e_b, (long)psiElems);
    conv_f32_bf16<<<2048, 256, 0, stream>>>(Psi_o, psi_o_b, (long)psiElems);
    conv_w1_pad  <<<2048, 256, 0, stream>>>(W1, w1b, (long)LAYERS * TWO_Z * K1PAD);
    conv_f32_bf16<<<2048, 256, 0, stream>>>(W2, w2b, (long)LAYERS * SOH * TWO_Z);
    conv_f32_bf16<<<2048, 256, 0, stream>>>(W3, w3b, (long)LAYERS * TWO_V * TWO_Z);
    // zero carries (x, v, xoh are contiguous in ws)
    zero_f32<<<1024, 256, 0, stream>>>(x_ws, (long)NBATCH * (TWO_S + TWO_V + SOH));

    for (int l = 0; l < LAYERS; ++l) {
        detnet_layerA<<<NBATCH, 128, 0, stream>>>(
            y_e, y_o, psi_e_b, psi_o_b, x_ws, v_ws, qv_b, d1, d2, d4, l);
        detnet_layerB<<<NBATCH / 32, 256, 0, stream>>>(
            qv_b,
            w1b + (size_t)l * TWO_Z * K1PAD, b1 + (size_t)l * TWO_Z,
            w2b + (size_t)l * SOH * TWO_Z,   b2 + (size_t)l * SOH,
            w3b + (size_t)l * TWO_V * TWO_Z, b3 + (size_t)l * TWO_V,
            xoh_ws, v_ws);
        detnet_layerC<<<NBATCH, 128, 0, stream>>>(
            xoh_ws, mapp_re, mapp_im, x_ws, out_xs, out_xoh, l);
    }
}
